// VAE_59270548685620
// MI455X (gfx1250) — matrix operand order repaired and checked by reading, not compiled, not run on hardware
//
#include <hip/hip_runtime.h>
#include <hip/hip_bf16.h>
#include <stdint.h>

typedef __attribute__((ext_vector_type(16))) __bf16 v16bf;
typedef __attribute__((ext_vector_type(8)))  float  v8f;

// ---------------- problem constants ----------------
static constexpr int BB  = 16;     // batch
static constexpr int NP  = 2048;   // points
static constexpr int PD  = 3;
static constexpr int ZD  = 128;
static constexpr int H1  = 128;
static constexpr int H2  = 256;
static constexpr int H3  = 512;
static constexpr int D1  = 512;
static constexpr int D2  = 1024;
static constexpr int D3  = NP * PD;        // 6144
static constexpr int MPTS = BB * NP;       // 32768 rows for encoder GEMMs

// ---------------- workspace layout (bytes) ----------------
static constexpr size_t WS_H1   = 0;                                   // 32768x128 bf16
static constexpr size_t WS_H2   = WS_H1   + (size_t)MPTS * H1 * 2;     // 32768x256 bf16
static constexpr size_t WS_WE2T = WS_H2   + (size_t)MPTS * H2 * 2;     // (256,128) bf16
static constexpr size_t WS_WE3T = WS_WE2T + (size_t)H2 * H1 * 2;       // (512,256) bf16
static constexpr size_t WS_WD1T = WS_WE3T + (size_t)H3 * H2 * 2;       // (512,128) bf16
static constexpr size_t WS_WD2T = WS_WD1T + (size_t)D1 * ZD * 2;       // (1024,512) bf16
static constexpr size_t WS_WD3T = WS_WD2T + (size_t)D2 * D1 * 2;       // (6144,1024) bf16
static constexpr size_t WS_GENC = WS_WD3T + (size_t)D3 * D2 * 2;       // 16x512 u32 (encoded max)
static constexpr size_t WS_Z    = WS_GENC + (size_t)BB * H3 * 4;       // 16x128 bf16
static constexpr size_t WS_HD1  = WS_Z    + (size_t)BB * ZD * 2;       // 16x512 bf16
static constexpr size_t WS_HD2  = WS_HD1  + (size_t)BB * D1 * 2;       // 16x1024 bf16
static constexpr size_t WS_Y    = WS_HD2  + (size_t)BB * D2 * 2;       // 16x6144 f32
static constexpr size_t WS_KL   = WS_Y    + (size_t)BB * D3 * 4;       // 16 f32
static constexpr size_t WS_CD   = WS_KL   + 64;                        // 16 f32
static constexpr size_t WS_END  = WS_CD   + 64;

// ---------------- device helpers ----------------
__device__ __forceinline__ unsigned f2mono(float f) {
  unsigned u = __float_as_uint(f);
  return (u & 0x80000000u) ? ~u : (u | 0x80000000u);
}
__device__ __forceinline__ float mono2f(unsigned u) {
  return (u & 0x80000000u) ? __uint_as_float(u & 0x7fffffffu) : __uint_as_float(~u);
}

__device__ __forceinline__ void wait_asynccnt0() {
#if __has_builtin(__builtin_amdgcn_s_wait_asynccnt)
  __builtin_amdgcn_s_wait_asynccnt(0);
#else
  asm volatile("s_wait_asynccnt 0x0" ::: "memory");
#endif
}

// issue one 16-byte async global->LDS DMA (ASYNCcnt-tracked)
__device__ __forceinline__ void async_copy_b128(unsigned lds_byte_addr,
                                                const void* gaddr) {
  asm volatile("global_load_async_to_lds_b128 %0, %1, off"
               :: "v"(lds_byte_addr),
                  "v"((unsigned long long)(uintptr_t)gaddr)
               : "memory");
}

// A fragment: 16x32 bf16, row = lane%16; lanes<16 hold K {k0..k0+7, k0+16..k0+23},
// lanes>=16 hold K {k0+8..k0+15, k0+24..k0+31}  (ISA 7.12.2 16-bit A layout)
__device__ __forceinline__ v16bf load_fragA(const __hip_bfloat16* rowptr, int k0, int lane) {
  int kh = ((lane >> 4) & 1) << 3;  // 0 or 8
  union { v16bf v; uint4 q[2]; } f;
  f.q[0] = *reinterpret_cast<const uint4*>(rowptr + k0 + kh);
  f.q[1] = *reinterpret_cast<const uint4*>(rowptr + k0 + kh + 16);
  return f.v;
}
// B fragment: 32x16 bf16, col = lane%16; lanes<16 hold K {k0..k0+7, k0+16..k0+23},
// lanes>=16 hold K {k0+8..k0+15, k0+24..k0+31}.  Wt is (N,K) row-major -> contiguous K per col.
__device__ __forceinline__ v16bf load_fragB(const __hip_bfloat16* rowptr, int k0, int lane) {
  int kh = ((lane >> 4) & 1) << 3;  // 0 or 8
  union { v16bf v; uint4 q[2]; } f;
  f.q[0] = *reinterpret_cast<const uint4*>(rowptr + k0 + kh);
  f.q[1] = *reinterpret_cast<const uint4*>(rowptr + k0 + kh + 16);
  return f.v;
}

// ---------------- kernels ----------------
__global__ void init_kernel(unsigned* gEnc, float* cd_b, int ng) {
  int i = blockIdx.x * blockDim.x + threadIdx.x;
  if (i < ng) gEnc[i] = f2mono(-3.4e38f);
  if (i < BB) cd_b[i] = 0.0f;
}

// Wt[n*K + k] = bf16(W[k*N + n])
__global__ void cvt_transpose_kernel(const float* __restrict__ W,
                                     __hip_bfloat16* __restrict__ Wt, int K, int N) {
  int idx = blockIdx.x * blockDim.x + threadIdx.x;
  if (idx >= K * N) return;
  int k = idx / N, n = idx - k * N;
  Wt[(size_t)n * K + k] = __float2bfloat16(W[(size_t)k * N + n]);
}

// encoder layer 1: 3 -> 128, relu, bf16 out (K=3: plain VALU)
__global__ __launch_bounds__(H1) void enc1_kernel(const float* __restrict__ x,
                                                  const float* __restrict__ We1,
                                                  const float* __restrict__ be1,
                                                  __hip_bfloat16* __restrict__ h1) {
  int p = blockIdx.x;          // point index (0..32767)
  int c = threadIdx.x;         // output channel
  float x0 = x[(size_t)p * 3 + 0];
  float x1 = x[(size_t)p * 3 + 1];
  float x2 = x[(size_t)p * 3 + 2];
  float v = fmaf(x0, We1[c], fmaf(x1, We1[H1 + c], fmaf(x2, We1[2 * H1 + c], be1[c])));
  h1[(size_t)p * H1 + c] = __float2bfloat16(fmaxf(v, 0.0f));
}

// stage shared A strip (16 rows x K, contiguous 32*K bytes) into LDS via async DMA
__device__ __forceinline__ void stage_A_tile(unsigned char* smem,
                                             const __hip_bfloat16* A, int m0, int K,
                                             int tid) {
  const unsigned char* gsrc = (const unsigned char*)(A + (size_t)m0 * K);
  unsigned lds0 = (unsigned)(uintptr_t)smem;
  int chunks = (16 * K * 2) >> 4;  // 16B chunks
  for (int c = tid; c < chunks; c += 256)
    async_copy_b128(lds0 + (unsigned)(c << 4), gsrc + ((size_t)c << 4));
  wait_asynccnt0();
  __syncthreads();
}

// generic bf16 WMMA GEMM: C(M,N) = act(A(M,K) @ Wt(N,K)^T + bias)
// 8 waves/WG; wave w owns a 16x16 tile at (blockIdx.y*16, blockIdx.x*128 + w*16).
// The 16-row A strip (shared by all 8 waves) is staged in LDS with async DMA.
template <bool RELU, bool OUTBF16>
__global__ __launch_bounds__(256) void wmma_gemm_kernel(
    const __hip_bfloat16* __restrict__ A, const __hip_bfloat16* __restrict__ Wt,
    const float* __restrict__ bias, void* __restrict__ Cout, int M, int N, int K) {
  extern __shared__ unsigned char smem[];  // 16*K*2 bytes
  int tid = threadIdx.x;
  int lane = tid & 31;
  int wave = tid >> 5;
  int m0 = blockIdx.y << 4;
  int n0 = (blockIdx.x << 7) + (wave << 4);

  stage_A_tile(smem, A, m0, K, tid);
  if (n0 >= N) return;  // uniform per wave (after barrier): EXEC stays all-1 for WMMA

  const __hip_bfloat16* arow = (const __hip_bfloat16*)smem + (size_t)(lane & 15) * K;
  const __hip_bfloat16* brow = Wt + (size_t)(n0 + (lane & 15)) * K;
  v8f acc = {0.f, 0.f, 0.f, 0.f, 0.f, 0.f, 0.f, 0.f};
  for (int k0 = 0; k0 < K; k0 += 32) {
    v16bf a = load_fragA(arow, k0, lane);   // ds_load_b128 x2
    v16bf b = load_fragB(brow, k0, lane);   // global_load_b128 x2
    acc = __builtin_amdgcn_wmma_f32_16x16x32_bf16(false, a, false, b, (short)0, acc,
                                                  false, false);
  }
  int ncol = n0 + (lane & 15);
  float bn = bias[ncol];
  int mbase = m0 + ((lane >> 4) << 3);  // +8 for upper half-wave
#pragma unroll
  for (int i = 0; i < 8; ++i) {
    float v = acc[i] + bn;
    if (RELU) v = fmaxf(v, 0.0f);
    size_t off = (size_t)(mbase + i) * N + ncol;
    if (OUTBF16)
      ((__hip_bfloat16*)Cout)[off] = __float2bfloat16(v);
    else
      ((float*)Cout)[off] = v;
  }
}

// encoder layer 3 GEMM fused with global max-pool over points (no h3 materialized)
__global__ __launch_bounds__(256) void wmma_gemm_maxpool_kernel(
    const __hip_bfloat16* __restrict__ A, const __hip_bfloat16* __restrict__ Wt,
    const float* __restrict__ bias, unsigned* __restrict__ gEnc, int M, int N, int K,
    int ptsPerBatch) {
  extern __shared__ unsigned char smem[];  // 16*K*2 bytes
  int tid = threadIdx.x;
  int lane = tid & 31;
  int wave = tid >> 5;
  int m0 = blockIdx.y << 4;
  int n0 = (blockIdx.x << 7) + (wave << 4);

  stage_A_tile(smem, A, m0, K, tid);
  if (n0 >= N) return;

  const __hip_bfloat16* arow = (const __hip_bfloat16*)smem + (size_t)(lane & 15) * K;
  const __hip_bfloat16* brow = Wt + (size_t)(n0 + (lane & 15)) * K;
  v8f acc = {0.f, 0.f, 0.f, 0.f, 0.f, 0.f, 0.f, 0.f};
  for (int k0 = 0; k0 < K; k0 += 32) {
    v16bf a = load_fragA(arow, k0, lane);
    v16bf b = load_fragB(brow, k0, lane);
    acc = __builtin_amdgcn_wmma_f32_16x16x32_bf16(false, a, false, b, (short)0, acc,
                                                  false, false);
  }
  int ncol = n0 + (lane & 15);
  float bn = bias[ncol];
  float vmax = -3.4e38f;
#pragma unroll
  for (int i = 0; i < 8; ++i) vmax = fmaxf(vmax, acc[i] + bn);
  // lanes l and l+16 hold the other 8 rows of the 16-row tile, same column
  vmax = fmaxf(vmax, __shfl_xor(vmax, 16, 32));
  if (lane < 16) {
    int batch = m0 / ptsPerBatch;  // 16-row tile never crosses batch (2048 % 16 == 0)
    atomicMax(&gEnc[(size_t)batch * N + ncol], f2mono(vmax));
  }
}

// latent heads + reparameterization + per-batch KL
__global__ __launch_bounds__(ZD) void latent_kernel(
    const unsigned* __restrict__ gEnc, const float* __restrict__ Wm,
    const float* __restrict__ bm, const float* __restrict__ Wv,
    const float* __restrict__ bv, const float* __restrict__ eps,
    __hip_bfloat16* __restrict__ zbf, float* __restrict__ kl_b) {
  __shared__ float g[H3];
  __shared__ float red[ZD];
  int b = blockIdx.x, c = threadIdx.x;
  for (int t = c; t < H3; t += ZD) g[t] = mono2f(gEnc[(size_t)b * H3 + t]);
  __syncthreads();
  float m = bm[c], v = bv[c];
  for (int k = 0; k < H3; ++k) {
    float gv = g[k];
    m = fmaf(gv, Wm[(size_t)k * ZD + c], m);
    v = fmaf(gv, Wv[(size_t)k * ZD + c], v);
  }
  float sp  = (v > 20.0f) ? v : log1pf(expf(v));   // softplus
  float var = sp + 1e-8f;
  float z   = m + sqrtf(var) * eps[(size_t)b * ZD + c];
  zbf[(size_t)b * ZD + c] = __float2bfloat16(z);
  red[c] = -logf(var) + var + m * m - 1.0f;
  __syncthreads();
  for (int s = ZD / 2; s > 0; s >>= 1) {
    if (c < s) red[c] += red[c + s];
    __syncthreads();
  }
  if (c == 0) kl_b[b] = 0.5f * red[0];
}

// one-directional Chamfer: for each a_i in Aset, min_j ||a_i - b_j||^2; mean over i
__global__ __launch_bounds__(256) void nnmin_kernel(
    const float* __restrict__ Aset, const float* __restrict__ Bset,
    float* __restrict__ cd_accum, int n, float scale) {
  __shared__ float bs[NP * 3];
  __shared__ float red[256];
  int b = blockIdx.y;
  int i = blockIdx.x * blockDim.x + threadIdx.x;
  const float* Bb = Bset + (size_t)b * n * 3;
  for (int t = threadIdx.x; t < n * 3; t += blockDim.x) bs[t] = Bb[t];
  __syncthreads();
  const float* Ab = Aset + (size_t)b * n * 3;
  float ax = Ab[(size_t)i * 3 + 0];
  float ay = Ab[(size_t)i * 3 + 1];
  float az = Ab[(size_t)i * 3 + 2];
  float mind = 3.4e38f;
  for (int j = 0; j < n; ++j) {
    float dx = ax - bs[3 * j + 0];
    float dy = ay - bs[3 * j + 1];
    float dz = az - bs[3 * j + 2];
    float d = fmaf(dx, dx, fmaf(dy, dy, dz * dz));
    mind = fminf(mind, d);
  }
  red[threadIdx.x] = mind;
  __syncthreads();
  for (int s = 128; s > 0; s >>= 1) {
    if (threadIdx.x < s) red[threadIdx.x] += red[threadIdx.x + s];
    __syncthreads();
  }
  if (threadIdx.x == 0) atomicAdd(&cd_accum[b], red[0] * scale);
}

__global__ void finalize_kernel(const float* __restrict__ kl_b,
                                const float* __restrict__ cd_b,
                                float* __restrict__ out) {
  if (threadIdx.x == 0 && blockIdx.x == 0) {
    float ks = 0.f, cs = 0.f;
    for (int i = 0; i < BB; ++i) { ks += kl_b[i]; cs += cd_b[i]; }
    float klm = ks / (float)BB, cdm = cs / (float)BB;
    out[0] = cdm + klm;  // nelbo
    out[1] = klm;        // kl_loss
    out[2] = cdm;        // x_reconst (chamfer)
  }
}

// ---------------- host launch ----------------
extern "C" void kernel_launch(void* const* d_in, const int* in_sizes, int n_in,
                              void* d_out, int out_size, void* d_ws, size_t ws_size,
                              hipStream_t stream) {
  (void)in_sizes; (void)n_in; (void)out_size;
  if (ws_size < WS_END) return;

  const float* x   = (const float*)d_in[0];
  const float* eps = (const float*)d_in[1];
  const float* We1 = (const float*)d_in[2];
  const float* be1 = (const float*)d_in[3];
  const float* We2 = (const float*)d_in[4];
  const float* be2 = (const float*)d_in[5];
  const float* We3 = (const float*)d_in[6];
  const float* be3 = (const float*)d_in[7];
  const float* Wm  = (const float*)d_in[8];
  const float* bm  = (const float*)d_in[9];
  const float* Wv  = (const float*)d_in[10];
  const float* bv  = (const float*)d_in[11];
  const float* Wd1 = (const float*)d_in[12];
  const float* bd1 = (const float*)d_in[13];
  const float* Wd2 = (const float*)d_in[14];
  const float* bd2 = (const float*)d_in[15];
  const float* Wd3 = (const float*)d_in[16];
  const float* bd3 = (const float*)d_in[17];

  char* ws = (char*)d_ws;
  __hip_bfloat16* h1   = (__hip_bfloat16*)(ws + WS_H1);
  __hip_bfloat16* h2   = (__hip_bfloat16*)(ws + WS_H2);
  __hip_bfloat16* We2t = (__hip_bfloat16*)(ws + WS_WE2T);
  __hip_bfloat16* We3t = (__hip_bfloat16*)(ws + WS_WE3T);
  __hip_bfloat16* Wd1t = (__hip_bfloat16*)(ws + WS_WD1T);
  __hip_bfloat16* Wd2t = (__hip_bfloat16*)(ws + WS_WD2T);
  __hip_bfloat16* Wd3t = (__hip_bfloat16*)(ws + WS_WD3T);
  unsigned*       gEnc = (unsigned*)(ws + WS_GENC);
  __hip_bfloat16* zbf  = (__hip_bfloat16*)(ws + WS_Z);
  __hip_bfloat16* hd1  = (__hip_bfloat16*)(ws + WS_HD1);
  __hip_bfloat16* hd2  = (__hip_bfloat16*)(ws + WS_HD2);
  float*          y    = (float*)(ws + WS_Y);
  float*          kl_b = (float*)(ws + WS_KL);
  float*          cd_b = (float*)(ws + WS_CD);
  float*          out  = (float*)d_out;

  // 0) per-call init (deterministic; atomically-updated buffers reset each call)
  init_kernel<<<dim3((BB * H3 + 255) / 256), 256, 0, stream>>>(gEnc, cd_b, BB * H3);

  // 1) weight convert + transpose to bf16 (N,K)
  auto cvt = [&](const float* W, __hip_bfloat16* Wt, int K, int N) {
    int tot = K * N;
    cvt_transpose_kernel<<<dim3((tot + 255) / 256), 256, 0, stream>>>(W, Wt, K, N);
  };
  cvt(We2, We2t, H1, H2);
  cvt(We3, We3t, H2, H3);
  cvt(Wd1, Wd1t, ZD, D1);
  cvt(Wd2, Wd2t, D1, D2);
  cvt(Wd3, Wd3t, D2, D3);

  // 2) encoder
  enc1_kernel<<<dim3(MPTS), H1, 0, stream>>>(x, We1, be1, h1);
  wmma_gemm_kernel<true, true>
      <<<dim3(H2 / 128, MPTS / 16), 256, (size_t)16 * H1 * 2, stream>>>(
          h1, We2t, be2, h2, MPTS, H2, H1);
  wmma_gemm_maxpool_kernel
      <<<dim3(H3 / 128, MPTS / 16), 256, (size_t)16 * H2 * 2, stream>>>(
          h2, We3t, be3, gEnc, MPTS, H3, H2, NP);

  // 3) latent heads + reparameterization + KL
  latent_kernel<<<dim3(BB), ZD, 0, stream>>>(gEnc, Wm, bm, Wv, bv, eps, zbf, kl_b);

  // 4) decoder
  wmma_gemm_kernel<true, true>
      <<<dim3(D1 / 128, 1), 256, (size_t)16 * ZD * 2, stream>>>(
          zbf, Wd1t, bd1, hd1, BB, D1, ZD);
  wmma_gemm_kernel<true, true>
      <<<dim3(D2 / 128, 1), 256, (size_t)16 * D1 * 2, stream>>>(
          hd1, Wd2t, bd2, hd2, BB, D2, D1);
  wmma_gemm_kernel<false, false>
      <<<dim3(D3 / 128, 1), 256, (size_t)16 * D2 * 2, stream>>>(
          hd2, Wd3t, bd3, y, BB, D3, D2);

  // 5) Chamfer (both directions), mean-of-min accumulated per batch
  nnmin_kernel<<<dim3(NP / 256, BB), 256, 0, stream>>>(y, x, cd_b, NP, 1.0f / (float)NP);
  nnmin_kernel<<<dim3(NP / 256, BB), 256, 0, stream>>>(x, y, cd_b, NP, 1.0f / (float)NP);

  // 6) scalars
  finalize_kernel<<<1, 32, 0, stream>>>(kl_b, cd_b, out);
}
